// GraphAttention_22316650070323
// MI455X (gfx1250) — matrix operand order repaired and checked by reading, not compiled, not run on hardware
//
#include <hip/hip_runtime.h>
#include <hip/hip_bf16.h>

// Problem sizes (fixed by reference)
#define NN    8192
#define FIN   512
#define FOUT  256
#define ALPHA 0.2f
#define JCHUNK (NN / 4)       // 4-way split of the attention K dimension

typedef __attribute__((ext_vector_type(16))) __bf16 v16bf;
typedef __attribute__((ext_vector_type(8)))  __bf16 v8bf;
typedef __attribute__((ext_vector_type(8)))  float  v8f;

union BF16x16 { v16bf v; v8bf h[2]; };

// ---------------------------------------------------------------------------
// Kernel 0: W [FIN][FOUT] f32  ->  WT [FOUT][FIN] bf16
// ---------------------------------------------------------------------------
__global__ __launch_bounds__(256)
void k_transpose_w(const float* __restrict__ W, __bf16* __restrict__ WT) {
    int idx = blockIdx.x * 256 + threadIdx.x;      // over FIN*FOUT
    int k = idx >> 8;                              // 0..511
    int n = idx & 255;                             // 0..255
    WT[(size_t)n * FIN + k] = (__bf16)W[(size_t)k * FOUT + n];
}

// ---------------------------------------------------------------------------
// Kernel 1: h = X @ W  via v_wmma_f32_16x16x32_bf16.
// One wave -> 16 rows x 128 cols (8 accumulator tiles); 2-way column split
// for occupancy. All 8 B tiles preloaded so WMMAs issue back-to-back.
// Also writes hT (bf16, [FOUT][NN]) for the attention pass.
// ---------------------------------------------------------------------------
__global__ __launch_bounds__(128)
void k_gemm_h(const float* __restrict__ X, const __bf16* __restrict__ WT,
              float* __restrict__ h, __bf16* __restrict__ hT) {
    const int wave = threadIdx.x >> 5;
    const int lane = threadIdx.x & 31;
    const int tile = blockIdx.x * 4 + wave;        // 0..1023
    const int rt   = tile >> 1;                    // row tile 0..511
    const int ch   = tile & 1;                     // column half
    const int ct0  = ch * 8;
    const int m0   = rt * 16;
    const int r    = lane & 15;
    const int hi   = lane >> 4;                    // half-wave id
    const int kb   = hi * 8;                       // A-layout K base (interleaved)
    const int kb2  = hi * 16;                      // unused; B loads use kb (interleaved)

    v8f acc[8];
#pragma unroll
    for (int c = 0; c < 8; ++c) acc[c] = (v8f)0.0f;

    const float* xr = X + (size_t)(m0 + r) * FIN;

    for (int k0 = 0; k0 < FIN; k0 += 32) {
        // ---- 1) preload all 8 B tiles (16 x b128 loads, one clause)
        BF16x16 B[8];
#pragma unroll
        for (int c = 0; c < 8; ++c) {
            const __bf16* bp = WT + (size_t)((ct0 + c) * 16 + r) * FIN + k0 + kb;
            B[c].h[0] = *(const v8bf*)bp;
            B[c].h[1] = *(const v8bf*)(bp + 16);
        }
        // ---- 2) A tile: X rows (f32 -> bf16), e <-> K = 16*(e>>3)+kb+(e&7)
        float4 a0 = *(const float4*)(xr + k0 + kb);
        float4 a1 = *(const float4*)(xr + k0 + kb + 4);
        float4 a2 = *(const float4*)(xr + k0 + 16 + kb);
        float4 a3 = *(const float4*)(xr + k0 + 16 + kb + 4);
        float av[16] = {a0.x,a0.y,a0.z,a0.w, a1.x,a1.y,a1.z,a1.w,
                        a2.x,a2.y,a2.z,a2.w, a3.x,a3.y,a3.z,a3.w};
        v16bf A;
#pragma unroll
        for (int e = 0; e < 16; ++e) A[e] = (__bf16)av[e];

        // ---- 3) WMMA chain, back-to-back
#pragma unroll
        for (int c = 0; c < 8; ++c)
            acc[c] = __builtin_amdgcn_wmma_f32_16x16x32_bf16(
                false, A, false, B[c].v, (short)0, acc[c], false, false);
    }

    // D layout: VGPR i holds M = i + 8*hi, N = r
#pragma unroll
    for (int c = 0; c < 8; ++c) {
        int n = (ct0 + c) * 16 + r;
#pragma unroll
        for (int i = 0; i < 8; ++i) {
            int m = m0 + hi * 8 + i;
            float v = acc[c][i];
            h[(size_t)m * FOUT + n] = v;
            hT[(size_t)n * NN + m]  = (__bf16)v;
        }
    }
}

// ---------------------------------------------------------------------------
// Kernel 2: f1 = h@a1, f2 = h@a2.  One wave per row.
// ---------------------------------------------------------------------------
__global__ __launch_bounds__(256)
void k_f12(const float* __restrict__ h, const float* __restrict__ a1,
           const float* __restrict__ a2, float* __restrict__ f1,
           float* __restrict__ f2) {
    const int wave = threadIdx.x >> 5;
    const int lane = threadIdx.x & 31;
    const int row  = blockIdx.x * 8 + wave;

    const float* hp = h + (size_t)row * FOUT + lane * 8;
    float4 x0 = *(const float4*)(hp);
    float4 x1 = *(const float4*)(hp + 4);
    float4 u0 = *(const float4*)(a1 + lane * 8);
    float4 u1 = *(const float4*)(a1 + lane * 8 + 4);
    float4 w0 = *(const float4*)(a2 + lane * 8);
    float4 w1 = *(const float4*)(a2 + lane * 8 + 4);

    float s1 = x0.x*u0.x + x0.y*u0.y + x0.z*u0.z + x0.w*u0.w
             + x1.x*u1.x + x1.y*u1.y + x1.z*u1.z + x1.w*u1.w;
    float s2 = x0.x*w0.x + x0.y*w0.y + x0.z*w0.z + x0.w*w0.w
             + x1.x*w1.x + x1.y*w1.y + x1.z*w1.z + x1.w*w1.w;

#pragma unroll
    for (int off = 16; off >= 1; off >>= 1) {
        s1 += __shfl_xor(s1, off, 32);
        s2 += __shfl_xor(s2, off, 32);
    }
    if (lane == 0) { f1[row] = s1; f2[row] = s2; }
}

// ---------------------------------------------------------------------------
// Kernel 3: per-row softmax stats (max, sum-of-exp) over masked LeakyReLU
// scores, online per lane + shfl combine. One coalesced sweep of adj.
// ---------------------------------------------------------------------------
__global__ __launch_bounds__(256)
void k_stats(const int* __restrict__ adj, const float* __restrict__ f1,
             const float* __restrict__ f2, float* __restrict__ mrow,
             float* __restrict__ lrow) {
    const int wave = threadIdx.x >> 5;
    const int lane = threadIdx.x & 31;
    const int row  = blockIdx.x * 8 + wave;

    const float fi = f1[row];
    const int* ar  = adj + (size_t)row * NN;

    float m = -1e30f, l = 0.f;
    for (int c0 = 0; c0 < NN; c0 += 128) {
        int c = c0 + lane * 4;
        int4   a = *(const int4*)(ar + c);
        float4 g = *(const float4*)(f2 + c);
        int   av[4] = {a.x, a.y, a.z, a.w};
        float gv[4] = {g.x, g.y, g.z, g.w};
#pragma unroll
        for (int t = 0; t < 4; ++t) {
            float e = fi + gv[t];
            e = (e >= 0.f) ? e : ALPHA * e;
            if (av[t] > 0) {
                if (e > m) { l = l * __expf(m - e) + 1.f; m = e; }
                else       { l += __expf(e - m); }
            }
        }
    }
#pragma unroll
    for (int off = 16; off >= 1; off >>= 1) {
        float m2 = __shfl_xor(m, off, 32);
        float l2 = __shfl_xor(l, off, 32);
        float mn = fmaxf(m, m2);
        l = l * __expf(m - mn) + l2 * __expf(m2 - mn);
        m = mn;
    }
    if (lane == 0) { mrow[row] = m; lrow[row] = (l > 0.f) ? l : 1.f; }
}

// ---------------------------------------------------------------------------
// Kernel 4: partial h' = softmax(e) @ h, flash-style. One wave owns
// (16 rows x 256 cols x one j-chunk). Per 32-col j-step:
//   1) preload all 16 B tiles (clause of 32 b128 loads)
//   2) build P tile in registers (adj/f2 loads + expf) while B is in flight
//   3) 16 back-to-back bf16 WMMAs
// adj is read exactly once across all waves. Partials summed by k_reduce_elu.
// ---------------------------------------------------------------------------
__device__ __forceinline__ float pfun(int a, float g, float fi, float mi,
                                      float invl) {
    float e = fi + g;
    e = (e >= 0.f) ? e : ALPHA * e;
    return (a > 0) ? __expf(e - mi) * invl : 0.f;
}

__global__ __launch_bounds__(128)
void k_attn_partial(const int* __restrict__ adj, const float* __restrict__ f1,
                    const float* __restrict__ f2, const float* __restrict__ mrow,
                    const float* __restrict__ lrow, const __bf16* __restrict__ hT,
                    float* __restrict__ partial) {   // [4][NN][FOUT]
    const int wave = threadIdx.x >> 5;
    const int lane = threadIdx.x & 31;
    const int tile = blockIdx.x * 4 + wave;   // 0..2047
    const int rt   = tile >> 2;               // row tile 0..511
    const int jc   = tile & 3;                // j chunk 0..3
    const int m0   = rt * 16;
    const int jbeg = jc * JCHUNK;
    const int r    = lane & 15;
    const int hi   = lane >> 4;
    const int kb   = hi * 8;                  // A K-interleave base
    const int kb2  = hi * 16;                 // unused; B loads use kb too

    const float fi   = f1[m0 + r];
    const float mi   = mrow[m0 + r];
    const float invl = 1.f / lrow[m0 + r];

    v8f acc[16];
#pragma unroll
    for (int ct = 0; ct < 16; ++ct) acc[ct] = (v8f)0.0f;

    const int* ar = adj + (size_t)(m0 + r) * NN;

    for (int j0 = jbeg; j0 < jbeg + JCHUNK; j0 += 32) {
        __builtin_prefetch(ar + j0 + 512, 0, 1);   // global_prefetch_b8

        // ---- 1) preload all 16 B tiles: hT[ct*16+r][j0+kb .. +7] and [j0+16+kb .. +7]
        BF16x16 B[16];
#pragma unroll
        for (int ct = 0; ct < 16; ++ct) {
            const __bf16* bp = hT + (size_t)(ct * 16 + r) * NN + j0 + kb;
            B[ct].h[0] = *(const v8bf*)bp;
            B[ct].h[1] = *(const v8bf*)(bp + 16);
        }

        // ---- 2) build P (A operand) while B loads are in flight
        int4 a0 = *(const int4*)(ar + j0 + kb);
        int4 a1 = *(const int4*)(ar + j0 + kb + 4);
        int4 a2 = *(const int4*)(ar + j0 + 16 + kb);
        int4 a3 = *(const int4*)(ar + j0 + 16 + kb + 4);
        float4 g0 = *(const float4*)(f2 + j0 + kb);
        float4 g1 = *(const float4*)(f2 + j0 + kb + 4);
        float4 g2 = *(const float4*)(f2 + j0 + 16 + kb);
        float4 g3 = *(const float4*)(f2 + j0 + 16 + kb + 4);

        int   aa[16] = {a0.x,a0.y,a0.z,a0.w, a1.x,a1.y,a1.z,a1.w,
                        a2.x,a2.y,a2.z,a2.w, a3.x,a3.y,a3.z,a3.w};
        float gg[16] = {g0.x,g0.y,g0.z,g0.w, g1.x,g1.y,g1.z,g1.w,
                        g2.x,g2.y,g2.z,g2.w, g3.x,g3.y,g3.z,g3.w};
        v16bf A;
#pragma unroll
        for (int e = 0; e < 16; ++e)
            A[e] = (__bf16)pfun(aa[e], gg[e], fi, mi, invl);

        // ---- 3) WMMA chain, back-to-back
#pragma unroll
        for (int ct = 0; ct < 16; ++ct)
            acc[ct] = __builtin_amdgcn_wmma_f32_16x16x32_bf16(
                false, A, false, B[ct].v, (short)0, acc[ct], false, false);
    }

    // Store partial (no ELU yet). D layout: VGPR i -> M = i + 8*hi, N = r.
    float* pout = partial + (size_t)jc * NN * FOUT;
#pragma unroll
    for (int ct = 0; ct < 16; ++ct) {
        int n = ct * 16 + r;
#pragma unroll
        for (int i = 0; i < 8; ++i) {
            int m = m0 + hi * 8 + i;
            pout[(size_t)m * FOUT + n] = acc[ct][i];
        }
    }
}

// ---------------------------------------------------------------------------
// Kernel 5: out = ELU(sum of 4 partials), vectorized float4.
// ---------------------------------------------------------------------------
__global__ __launch_bounds__(256)
void k_reduce_elu(const float* __restrict__ partial, float* __restrict__ out) {
    size_t base = ((size_t)blockIdx.x * 256 + threadIdx.x) * 4;  // over NN*FOUT
    const size_t stride = (size_t)NN * FOUT;
    float4 s0 = *(const float4*)(partial + base);
    float4 s1 = *(const float4*)(partial + stride + base);
    float4 s2 = *(const float4*)(partial + 2 * stride + base);
    float4 s3 = *(const float4*)(partial + 3 * stride + base);
    float v[4] = {s0.x + s1.x + s2.x + s3.x,
                  s0.y + s1.y + s2.y + s3.y,
                  s0.z + s1.z + s2.z + s3.z,
                  s0.w + s1.w + s2.w + s3.w};
    float4 o;
    o.x = (v[0] > 0.f) ? v[0] : (__expf(v[0]) - 1.f);
    o.y = (v[1] > 0.f) ? v[1] : (__expf(v[1]) - 1.f);
    o.z = (v[2] > 0.f) ? v[2] : (__expf(v[2]) - 1.f);
    o.w = (v[3] > 0.f) ? v[3] : (__expf(v[3]) - 1.f);
    *(float4*)(out + base) = o;
}

// ---------------------------------------------------------------------------
extern "C" void kernel_launch(void* const* d_in, const int* in_sizes, int n_in,
                              void* d_out, int out_size, void* d_ws,
                              size_t ws_size, hipStream_t stream) {
    const float* X   = (const float*)d_in[0];   // [NN, FIN]
    const int*   adj = (const int*)d_in[1];     // [NN, NN]
    const float* W   = (const float*)d_in[2];   // [FIN, FOUT]
    const float* a1  = (const float*)d_in[3];   // [FOUT, 1]
    const float* a2  = (const float*)d_in[4];   // [FOUT, 1]
    float* out = (float*)d_out;                 // [NN, FOUT]

    char* ws = (char*)d_ws;
    size_t off = 0;
    __bf16* WT   = (__bf16*)(ws + off); off += (size_t)FOUT * FIN * 2;      // 256 KB
    float*  h    = (float*) (ws + off); off += (size_t)NN * FOUT * 4;       // 8 MB
    __bf16* hT   = (__bf16*)(ws + off); off += (size_t)FOUT * NN * 2;       // 4 MB
    float*  f1   = (float*) (ws + off); off += (size_t)NN * 4;
    float*  f2   = (float*) (ws + off); off += (size_t)NN * 4;
    float*  mrw  = (float*) (ws + off); off += (size_t)NN * 4;
    float*  lrw  = (float*) (ws + off); off += (size_t)NN * 4;
    float*  part = (float*) (ws + off); off += (size_t)4 * NN * FOUT * 4;   // 32 MB

    k_transpose_w<<<(FIN * FOUT) / 256, 256, 0, stream>>>(W, WT);
    k_gemm_h<<<(NN / 16) * 2 / 4, 128, 0, stream>>>(X, WT, h, hT);
    k_f12<<<NN / 8, 256, 0, stream>>>(h, a1, a2, f1, f2);
    k_stats<<<NN / 8, 256, 0, stream>>>(adj, f1, f2, mrw, lrw);
    k_attn_partial<<<(NN / 16) * 4 / 4, 128, 0, stream>>>(adj, f1, f2, mrw,
                                                          lrw, hT, part);
    k_reduce_elu<<<(NN * FOUT) / (256 * 4), 256, 0, stream>>>(part, out);
}
